// MQGoraTransformer_14035953123838
// MI455X (gfx1250) — matrix operand order repaired and checked by reading, not compiled, not run on hardware
//
#include <hip/hip_runtime.h>
#include <hip/hip_bf16.h>
#include <math.h>

// Problem constants
#define B_   2048
#define K_   64
#define M_   8
#define D_   64
#define H_   4
#define L_   2
#define FF_  128
#define NF_  32
#define OBS_ 32
#define OUT_ 8
#define DH_  16
#define EPS_ 1e-6f

#if __has_builtin(__builtin_amdgcn_global_load_async_to_lds_b128)
#define HAVE_ASYNC_LDS 1
#else
#define HAVE_ASYNC_LDS 0
#endif

typedef __attribute__((ext_vector_type(16))) _Float16 half16;
typedef __attribute__((ext_vector_type(8)))  float    f32x8;
typedef __attribute__((ext_vector_type(4)))  float    f32x4;
typedef __attribute__((ext_vector_type(4)))  int      i32x4;

__device__ __forceinline__ f32x8 wmma_f16(half16 a, half16 b, f32x8 c) {
  // D = A(16x32,f16) * B(32x16,f16) + C(16x16,f32)
  return __builtin_amdgcn_wmma_f32_16x16x32_f16(false, a, false, b, (short)0, c,
                                                false, false);
}

// A-fragment (16x32, f16) from row-major LDS tile, vectorized:
// lane L holds row M=L&15; halves j=0..7 are K=k0+hh+0..7 (16 contiguous bytes),
// j=8..15 are K=k0+hh+16..23 (16 contiguous bytes). hh = (L>>4)*8.
__device__ __forceinline__ half16 frag_a(const _Float16* s, int ld, int r0, int k0, int lane) {
  const int r  = r0 + (lane & 15);
  const int hh = (lane >> 4) * 8;
  const _Float16* p = s + r * ld + k0 + hh;
  union { f32x4 v[2]; half16 h; } u;
  u.v[0] = *(const f32x4*)(p);       // 1x ds_load_b128
  u.v[1] = *(const f32x4*)(p + 16);  // 1x ds_load_b128
  return u.h;
}

// B operand staged pre-swizzled in LDS: chunk c = tile*32 + lane holds the 16
// halfwords of that lane's fragment contiguously, so a fragment is two
// ds_load_b128. Swizzle: element j of chunk (tile=kt*ntiles+nt, lane) is
// W[(kt*32 + ((lane>>4)<<3) + j + (j>=8?8:0)) * N + nt*16 + (lane&15)].
__device__ __forceinline__ half16 frag_b_swz(const _Float16* s, int tile, int lane) {
  const _Float16* p = s + (tile * 32 + lane) * 16;
  union { f32x4 v[2]; half16 h; } u;
  u.v[0] = *(const f32x4*)(p);
  u.v[1] = *(const f32x4*)(p + 8);
  return u.h;
}

__device__ __forceinline__ void stage_b_swz(_Float16* dst, const float* __restrict__ W,
                                            int N, int nchunks, int tid, int nth) {
  const int ntiles = N >> 4;
  for (int c = tid; c < nchunks; c += nth) {
    const int ln    = c & 31;
    const int tile  = c >> 5;
    const int kt    = tile / ntiles;
    const int nt    = tile - kt * ntiles;
    const int kbase = kt * 32 + ((ln >> 4) << 3);
    const int n     = nt * 16 + (ln & 15);
    _Float16* d = dst + c * 16;
#pragma unroll
    for (int j = 0; j < 16; ++j) d[j] = (_Float16)W[(kbase + j + ((j >> 3) << 3)) * N + n];
  }
}

__device__ __forceinline__ float gelu_tanh(float x) {
  const float c = 0.7978845608028654f; // sqrt(2/pi)
  return 0.5f * x * (1.0f + tanhf(c * (x + 0.044715f * x * x * x)));
}

// ---------------------------------------------------------------------------
// 1) h = x_anc @ W_anc + b_anc            (2048x32 @ 32x64, WMMA)
// ---------------------------------------------------------------------------
__global__ __launch_bounds__(64) void k_anchor_gemm(const float* __restrict__ x_anc,
                                                    const float* __restrict__ W_anc,
                                                    const float* __restrict__ b_anc,
                                                    float* __restrict__ h_raw) {
  __shared__ alignas(16) _Float16 sA[16 * NF_];
  __shared__ alignas(16) _Float16 sW[NF_ * D_];   // swizzled, 4 tiles (kt=0, nt=0..3)
  const int row0 = blockIdx.x * 16;
  const int tid  = threadIdx.x;
  const int w    = tid >> 5;
  const int lane = tid & 31;
#pragma unroll
  for (int i = 0; i < 8; ++i) {  // 512 elements of A
    const int idx = tid + i * 64;
    sA[idx] = (_Float16)x_anc[row0 * NF_ + idx];
  }
  stage_b_swz(sW, W_anc, D_, 128, tid, 64);
  __syncthreads();
  const half16 a = frag_a(sA, NF_, 0, 0, lane);
#pragma unroll
  for (int nt = 0; nt < 2; ++nt) {
    const int n0 = w * 32 + nt * 16;
    f32x8 acc = {};
    const half16 bb = frag_b_swz(sW, n0 >> 4, lane);
    acc = wmma_f16(a, bb, acc);
    const int col = n0 + (lane & 15);
#pragma unroll
    for (int i = 0; i < 8; ++i) {
      const int r = i + 8 * (lane >> 4);
      h_raw[(row0 + r) * D_ + col] = acc[i] + b_anc[col];
    }
  }
}

// ---------------------------------------------------------------------------
// 2) batch-norm stats over B -> per-column scale/shift
// ---------------------------------------------------------------------------
__global__ __launch_bounds__(64) void k_bnstats(const float* __restrict__ h_raw,
                                                const float* __restrict__ bn_g,
                                                const float* __restrict__ bn_b,
                                                float* __restrict__ ss) {
  const int d = threadIdx.x;
  float s = 0.f, q = 0.f;
  for (int b = 0; b < B_; ++b) {
    const float v = h_raw[b * D_ + d];
    s += v; q += v * v;
  }
  const float mu  = s * (1.0f / B_);
  const float var = q * (1.0f / B_) - mu * mu;
  const float sc  = bn_g[d] * rsqrtf(var + 1e-5f);
  ss[d]       = sc;
  ss[D_ + d]  = bn_b[d] - mu * sc;
}

__global__ __launch_bounds__(256) void k_bnapply(const float* __restrict__ h_raw,
                                                 const float* __restrict__ ss,
                                                 float* __restrict__ h_anc) {
  const int i = blockIdx.x * 256 + threadIdx.x;
  const int d = i & (D_ - 1);
  h_anc[i] = h_raw[i] * ss[d] + ss[D_ + d];
}

// ---------------------------------------------------------------------------
// 3) prep: qk = Wk_read @ q_read ; tau = softplus(tau_raw)
// ---------------------------------------------------------------------------
__global__ __launch_bounds__(64) void k_prep(const float* __restrict__ Wk_read,
                                             const float* __restrict__ q_read,
                                             const float* __restrict__ tau_raw,
                                             float* __restrict__ qk,
                                             float* __restrict__ tau_ws,
                                             float* __restrict__ tau_out) {
  const int d = threadIdx.x;
  float acc = 0.f;
  for (int e = 0; e < D_; ++e) acc += Wk_read[d * D_ + e] * q_read[e];
  qk[d] = acc;
  if (d < H_) {
    const float t = logf(1.0f + expf(tau_raw[d]));
    tau_ws[d]  = t;
    tau_out[d] = t;
  }
}

// ---------------------------------------------------------------------------
// 4) fused view attention: per (b) block, wave m handles view m.
//    hn = f16(x_nei[b]) @ f16(W_view[m]) + b_view[m]        (WMMA)
//    sc = hn . qk / 8 + log(ew*mask + eps); softmax over K
//    ctx_m = (attn^T @ hn) @ Wv_read ; ctx = mean_m ctx_m
//    h_shared (m==0 tile) stored as f16.
// ---------------------------------------------------------------------------
__global__ __launch_bounds__(256) void k_view_attn(const float* __restrict__ x_nei,
                                                   const float* __restrict__ ew,
                                                   const float* __restrict__ vmsk,
                                                   const float* __restrict__ W_view,
                                                   const float* __restrict__ b_view,
                                                   const float* __restrict__ Wv_read,
                                                   const float* __restrict__ qk,
                                                   float* __restrict__ ctx,
                                                   _Float16* __restrict__ hs_out) {
  __shared__ alignas(16) _Float16 sX[K_ * NF_];       // 4 KB
  __shared__ alignas(16) _Float16 sWs[M_][NF_ * D_];  // 32 KB (swizzled per view)
  __shared__ alignas(16) _Float16 sHn[M_][K_ * D_];   // 64 KB
  __shared__ float sAttn[M_][K_];
  __shared__ float sU[M_][D_];
  __shared__ float sCtx[M_][D_];
  __shared__ float sQk[D_];

  const int b    = blockIdx.x;
  const int tid  = threadIdx.x;
  const int m    = tid >> 5;
  const int lane = tid & 31;

#pragma unroll
  for (int i = 0; i < 8; ++i) {  // 2048 floats of x_nei[b]
    const int idx = tid + i * 256;
    sX[idx] = (_Float16)x_nei[b * (K_ * NF_) + idx];
  }
  if (tid < D_) sQk[tid] = qk[tid];
  // each wave stages its own W_view[m] (32x64) pre-swizzled: 128 chunks
  stage_b_swz(&sWs[m][0], W_view + m * (NF_ * D_), D_, 128, lane, 32);
  __syncthreads();

  // hn (64x64) via 4x4 WMMA tiles of 16x16, K=32 in one step
#pragma unroll
  for (int rt = 0; rt < 4; ++rt) {
    const half16 a = frag_a(sX, NF_, rt * 16, 0, lane);
#pragma unroll
    for (int nt = 0; nt < 4; ++nt) {
      f32x8 acc = {};
      const half16 bb = frag_b_swz(&sWs[m][0], nt, lane);
      acc = wmma_f16(a, bb, acc);
      const int col = nt * 16 + (lane & 15);
#pragma unroll
      for (int i = 0; i < 8; ++i) {
        const int r = rt * 16 + i + 8 * (lane >> 4);
        const float v = acc[i] + b_view[m * D_ + col];
        const _Float16 hv = (_Float16)v;
        sHn[m][r * D_ + col] = hv;
        if (m == 0) hs_out[b * (K_ * D_) + r * D_ + col] = hv;
      }
    }
  }
  __syncthreads();

  // scores + softmax over K (2 rows per lane)
  float sc0, sc1;
  {
    float s[2];
#pragma unroll
    for (int t = 0; t < 2; ++t) {
      const int k = lane + t * 32;
      float dotv = 0.f;
      for (int d = 0; d < D_; ++d) dotv += (float)sHn[m][k * D_ + d] * sQk[d];
      const int wi = (b * K_ + k) * M_ + m;
      const float w = ew[wi] * vmsk[wi];
      s[t] = dotv * 0.125f + logf(w + EPS_);
    }
    sc0 = s[0]; sc1 = s[1];
  }
  float mx = fmaxf(sc0, sc1);
#pragma unroll
  for (int off = 16; off >= 1; off >>= 1) mx = fmaxf(mx, __shfl_xor(mx, off, 32));
  const float e0 = expf(sc0 - mx), e1 = expf(sc1 - mx);
  float ssum = e0 + e1;
#pragma unroll
  for (int off = 16; off >= 1; off >>= 1) ssum += __shfl_xor(ssum, off, 32);
  const float inv = 1.0f / ssum;
  sAttn[m][lane]      = e0 * inv;
  sAttn[m][lane + 32] = e1 * inv;
  __syncthreads();

  // u = attn^T @ hn  (2 columns per lane)
#pragma unroll
  for (int t = 0; t < 2; ++t) {
    const int c = lane + t * 32;
    float u = 0.f;
    for (int k = 0; k < K_; ++k) u += sAttn[m][k] * (float)sHn[m][k * D_ + c];
    sU[m][c] = u;
  }
  __syncthreads();

  // ctx_m = u @ Wv_read (2 outputs per lane)
#pragma unroll
  for (int t = 0; t < 2; ++t) {
    const int d = lane + t * 32;
    float cv = 0.f;
    for (int c = 0; c < D_; ++c) cv += sU[m][c] * Wv_read[c * D_ + d];
    sCtx[m][d] = cv;
  }
  __syncthreads();

  if (tid < D_) {
    float acc = 0.f;
#pragma unroll
    for (int mm = 0; mm < M_; ++mm) acc += sCtx[mm][tid];
    ctx[b * D_ + tid] = acc * 0.125f;
  }
}

// ---------------------------------------------------------------------------
// 5) gating MLP + pi softmax + log-gate table
// ---------------------------------------------------------------------------
__global__ __launch_bounds__(64) void k_gate(const float* __restrict__ g_anc,
                                             const float* __restrict__ ctx,
                                             const float* __restrict__ ew,
                                             const float* __restrict__ vmsk,
                                             const float* __restrict__ Wr1,
                                             const float* __restrict__ br1,
                                             const float* __restrict__ Wr2,
                                             const float* __restrict__ br2,
                                             float* __restrict__ pi_out,
                                             float* __restrict__ lg_ws) {
  __shared__ float sR[64];
  __shared__ float sPi[H_ * M_];
  const int b = blockIdx.x;
  const int t = threadIdx.x;

  float acc = br1[t];
  for (int j = 0; j < OBS_; ++j) acc += g_anc[b * OBS_ + j] * Wr1[j * 64 + t];
  for (int j = 0; j < D_; ++j)   acc += ctx[b * D_ + j] * Wr1[(OBS_ + j) * 64 + t];
  sR[t] = fmaxf(acc, 0.f);
  __syncthreads();

  if (t < H_ * M_) {  // threads 0..31 = wave 0; groups of 8 share one head
    float lg = br2[t];
    for (int j = 0; j < 64; ++j) lg += sR[j] * Wr2[j * (H_ * M_) + t];
    float mx = lg;
#pragma unroll
    for (int off = 1; off < 8; off <<= 1) mx = fmaxf(mx, __shfl_xor(mx, off, 32));
    const float e = expf(lg - mx);
    float s = e;
#pragma unroll
    for (int off = 1; off < 8; off <<= 1) s += __shfl_xor(s, off, 32);
    const float p = e / s;
    sPi[t] = p;
    pi_out[b * (H_ * M_) + t] = p;
  }
  __syncthreads();

#pragma unroll
  for (int i = 0; i < 4; ++i) {
    const int idx = i * 64 + t;           // idx over H*K = 256
    const int h = idx >> 6, k = idx & 63;
    float g = 0.f;
#pragma unroll
    for (int mm = 0; mm < M_; ++mm) {
      const int wi = (b * K_ + k) * M_ + mm;
      g += sPi[h * M_ + mm] * ew[wi] * vmsk[wi];
    }
    lg_ws[b * (H_ * K_) + idx] = logf(g + EPS_);
  }
}

// ---------------------------------------------------------------------------
// 6) per-layer MQ attention (projections commuted to per-row vector ops)
// ---------------------------------------------------------------------------
__global__ __launch_bounds__(64) void k_attn_layer(int l,
                                                   const _Float16* __restrict__ hs,
                                                   const float* __restrict__ lg_ws,
                                                   const float* __restrict__ Wq,
                                                   const float* __restrict__ Wk,
                                                   const float* __restrict__ Wv,
                                                   const float* __restrict__ Wo,
                                                   const float* __restrict__ ln1_g,
                                                   const float* __restrict__ ln1_b,
                                                   const float* __restrict__ tau,
                                                   float* __restrict__ h_anc) {
  __shared__ float    sH[D_];
  __shared__ alignas(16) _Float16 sHs[K_ * D_];
  __shared__ float    sQ[D_];
  __shared__ float    sWq[H_ * D_];
  __shared__ float    sS[H_ * K_];
  __shared__ float    sU[H_ * D_];
  __shared__ float    sO[D_];
  __shared__ float    sRed[D_];
  const int b = blockIdx.x;
  const int t = threadIdx.x;
  const int lw = l * D_ * D_;

  __builtin_prefetch(Wq + lw + t * 16, 0, 3);
  __builtin_prefetch(Wk + lw + t * 16, 0, 3);

  sH[t] = h_anc[b * D_ + t];
#if HAVE_ASYNC_LDS
  {
    // h_shared[b]: 8 KB f16, copied verbatim -> async DMA to LDS (ASYNCcnt)
    const _Float16* gsrc = hs + (size_t)b * (K_ * D_);
#pragma unroll
    for (int i = 0; i < 8; ++i) {
      const int off = (t + i * 64) * 8;  // 16 bytes per op per lane
      __builtin_amdgcn_global_load_async_to_lds_b128(
          (__attribute__((address_space(1))) i32x4*)(gsrc + off),
          (__attribute__((address_space(3))) i32x4*)(sHs + off), 0, 0);
    }
    asm volatile("s_wait_asynccnt 0x0" ::: "memory");
  }
#else
  for (int i = 0; i < K_; ++i) sHs[i * D_ + t] = hs[(size_t)b * (K_ * D_) + i * D_ + t];
#endif
  __syncthreads();

  {  // q = h_anc @ Wq[l]
    float qv = 0.f;
    for (int c = 0; c < D_; ++c) qv += sH[c] * Wq[lw + c * D_ + t];
    sQ[t] = qv;
  }
  __syncthreads();

#pragma unroll
  for (int i = 0; i < 4; ++i) {  // wq[h][c] = Wk[:,hDH:] . q[h]
    const int idx = i * 64 + t;
    const int h = idx >> 6, c = idx & 63;
    float a = 0.f;
#pragma unroll
    for (int d = 0; d < DH_; ++d) a += Wk[lw + c * D_ + h * DH_ + d] * sQ[h * DH_ + d];
    sWq[idx] = a;
  }
  __syncthreads();

#pragma unroll
  for (int i = 0; i < 4; ++i) {  // s[h,k]
    const int idx = i * 64 + t;
    const int h = idx >> 6, k = idx & 63;
    float a = 0.f;
    for (int c = 0; c < D_; ++c) a += (float)sHs[k * D_ + c] * sWq[h * D_ + c];
    sS[idx] = a / (4.0f * tau[h]) + lg_ws[b * (H_ * K_) + idx];
  }
  __syncthreads();

  if (t < H_) {  // softmax over K per head
    float mx = -3.4e38f;
    for (int k = 0; k < K_; ++k) mx = fmaxf(mx, sS[t * K_ + k]);
    float s = 0.f;
    for (int k = 0; k < K_; ++k) { const float e = expf(sS[t * K_ + k] - mx); sS[t * K_ + k] = e; s += e; }
    const float inv = 1.0f / s;
    for (int k = 0; k < K_; ++k) sS[t * K_ + k] *= inv;
  }
  __syncthreads();

#pragma unroll
  for (int i = 0; i < 4; ++i) {  // uh[h,c] = a[h,:] @ hs
    const int idx = i * 64 + t;
    const int h = idx >> 6, c = idx & 63;
    float a = 0.f;
    for (int k = 0; k < K_; ++k) a += sS[h * K_ + k] * (float)sHs[k * D_ + c];
    sU[idx] = a;
  }
  __syncthreads();

  {  // o[e] = uh[h] @ Wv[:, e]
    const int h = t >> 4;
    float o = 0.f;
    for (int c = 0; c < D_; ++c) o += sU[h * D_ + c] * Wv[lw + c * D_ + t];
    sO[t] = o;
  }
  __syncthreads();

  float v = sH[t];
  {  // o @ Wo, residual
    float oc = 0.f;
    for (int e = 0; e < D_; ++e) oc += sO[e] * Wo[lw + e * D_ + t];
    v += oc;
  }
  sRed[t] = v;
  __syncthreads();
  float mu = 0.f;
  for (int j = 0; j < D_; ++j) mu += sRed[j];
  mu *= (1.0f / D_);
  float var = 0.f;
  for (int j = 0; j < D_; ++j) { const float d = sRed[j] - mu; var += d * d; }
  var *= (1.0f / D_);
  h_anc[b * D_ + t] = (v - mu) * rsqrtf(var + 1e-5f) * ln1_g[l * D_ + t] + ln1_b[l * D_ + t];
}

// ---------------------------------------------------------------------------
// 7) per-layer FFN: gelu(h@W1+b1)@W2+b2, residual + LN    (WMMA, 16-row tiles)
// ---------------------------------------------------------------------------
__global__ __launch_bounds__(128) void k_ffn_layer(int l,
                                                   const float* __restrict__ W1,
                                                   const float* __restrict__ b1,
                                                   const float* __restrict__ W2,
                                                   const float* __restrict__ b2,
                                                   const float* __restrict__ ln2_g,
                                                   const float* __restrict__ ln2_b,
                                                   float* __restrict__ h_anc) {
  __shared__ alignas(16) _Float16 sA[16 * D_];     // 2 KB
  __shared__ float    sRes[16 * D_];               // 4 KB
  __shared__ alignas(16) _Float16 sW1s[D_ * FF_];  // 16 KB (swizzled)
  __shared__ alignas(16) _Float16 sF[16 * FF_];    // 4 KB
  __shared__ alignas(16) _Float16 sW2s[FF_ * D_];  // 16 KB (swizzled)
  __shared__ float    sOut[16 * D_];               // 4 KB
  const int row0 = blockIdx.x * 16;
  const int tid  = threadIdx.x;
  const int w    = tid >> 5;
  const int lane = tid & 31;

  __builtin_prefetch(W1 + l * D_ * FF_ + tid * 64, 0, 3);
  __builtin_prefetch(W2 + l * FF_ * D_ + tid * 64, 0, 3);

#pragma unroll
  for (int i = 0; i < 8; ++i) {     // 1024 activations
    const int idx = tid + i * 128;
    const float v = h_anc[row0 * D_ + idx];
    sA[idx] = (_Float16)v;
    sRes[idx] = v;
  }
  stage_b_swz(sW1s, W1 + l * D_ * FF_, FF_, 512, tid, 128);  // 64x128
  stage_b_swz(sW2s, W2 + l * FF_ * D_, D_, 512, tid, 128);   // 128x64
  __syncthreads();

  // GEMM1: 16x64 @ 64x128, wave w owns columns [w*32, w*32+32)
#pragma unroll
  for (int nn = 0; nn < 2; ++nn) {
    const int n0 = w * 32 + nn * 16;
    const int nt = n0 >> 4;
    f32x8 acc = {};
#pragma unroll
    for (int kt = 0; kt < 2; ++kt) {
      const half16 a  = frag_a(sA, D_, 0, kt * 32, lane);
      const half16 bb = frag_b_swz(sW1s, kt * 8 + nt, lane);
      acc = wmma_f16(a, bb, acc);
    }
    const int col = n0 + (lane & 15);
#pragma unroll
    for (int i = 0; i < 8; ++i) {
      const int r = i + 8 * (lane >> 4);
      sF[r * FF_ + col] = (_Float16)gelu_tanh(acc[i] + b1[l * FF_ + col]);
    }
  }
  __syncthreads();

  // GEMM2: 16x128 @ 128x64, wave w owns columns [w*16, w*16+16)
  {
    f32x8 acc = {};
#pragma unroll
    for (int kt = 0; kt < 4; ++kt) {
      const half16 a  = frag_a(sF, FF_, 0, kt * 32, lane);
      const half16 bb = frag_b_swz(sW2s, kt * 4 + w, lane);
      acc = wmma_f16(a, bb, acc);
    }
    const int col = w * 16 + (lane & 15);
#pragma unroll
    for (int i = 0; i < 8; ++i) {
      const int r = i + 8 * (lane >> 4);
      sOut[r * D_ + col] = acc[i] + b2[l * D_ + col] + sRes[r * D_ + col];
    }
  }
  __syncthreads();

  if (tid < 16) {  // row-wise LayerNorm
    const int r = tid;
    float mu = 0.f;
    for (int d = 0; d < D_; ++d) mu += sOut[r * D_ + d];
    mu *= (1.0f / D_);
    float var = 0.f;
    for (int d = 0; d < D_; ++d) { const float x = sOut[r * D_ + d] - mu; var += x * x; }
    var *= (1.0f / D_);
    const float inv = rsqrtf(var + 1e-5f);
    for (int d = 0; d < D_; ++d)
      h_anc[(row0 + r) * D_ + d] =
          (sOut[r * D_ + d] - mu) * inv * ln2_g[l * D_ + d] + ln2_b[l * D_ + d];
  }
}

// ---------------------------------------------------------------------------
// 8) head: z = LN(h_anc); pred = gelu(z@Wp1+bp1)@Wp2+bp2
// ---------------------------------------------------------------------------
__global__ __launch_bounds__(64) void k_head(const float* __restrict__ h_anc,
                                             const float* __restrict__ hg,
                                             const float* __restrict__ hb,
                                             const float* __restrict__ Wp1,
                                             const float* __restrict__ bp1,
                                             const float* __restrict__ Wp2,
                                             const float* __restrict__ bp2,
                                             float* __restrict__ pred) {
  __shared__ float sRed[D_];
  __shared__ float sZ[D_];
  __shared__ float sG[D_ / 2];
  const int b = blockIdx.x;
  const int t = threadIdx.x;
  const float v = h_anc[b * D_ + t];
  sRed[t] = v;
  __syncthreads();
  float mu = 0.f;
  for (int j = 0; j < D_; ++j) mu += sRed[j];
  mu *= (1.0f / D_);
  float var = 0.f;
  for (int j = 0; j < D_; ++j) { const float d = sRed[j] - mu; var += d * d; }
  var *= (1.0f / D_);
  sZ[t] = (v - mu) * rsqrtf(var + 1e-5f) * hg[t] + hb[t];
  __syncthreads();
  if (t < D_ / 2) {
    float a = bp1[t];
    for (int c = 0; c < D_; ++c) a += sZ[c] * Wp1[c * (D_ / 2) + t];
    sG[t] = gelu_tanh(a);
  }
  __syncthreads();
  if (t < OUT_) {
    float a = bp2[t];
    for (int j = 0; j < D_ / 2; ++j) a += sG[j] * Wp2[j * OUT_ + t];
    pred[b * OUT_ + t] = a;
  }
}

// ---------------------------------------------------------------------------
extern "C" void kernel_launch(void* const* d_in, const int* in_sizes, int n_in,
                              void* d_out, int out_size, void* d_ws, size_t ws_size,
                              hipStream_t stream) {
  (void)in_sizes; (void)n_in; (void)out_size; (void)ws_size;
  const float* x_anc = (const float*)d_in[0];
  const float* g_anc = (const float*)d_in[1];
  const float* x_nei = (const float*)d_in[2];
  const float* ew    = (const float*)d_in[3];
  const float* vmsk  = (const float*)d_in[4];
  const float* W_anc   = (const float*)d_in[5];
  const float* b_anc   = (const float*)d_in[6];
  const float* bn_g    = (const float*)d_in[7];
  const float* bn_b    = (const float*)d_in[8];
  const float* W_view  = (const float*)d_in[9];
  const float* b_view  = (const float*)d_in[10];
  const float* Wk_read = (const float*)d_in[11];
  const float* Wv_read = (const float*)d_in[12];
  const float* q_read  = (const float*)d_in[13];
  const float* Wr1     = (const float*)d_in[14];
  const float* br1     = (const float*)d_in[15];
  const float* Wr2     = (const float*)d_in[16];
  const float* br2     = (const float*)d_in[17];
  const float* tau_raw = (const float*)d_in[18];
  const float* Wq      = (const float*)d_in[19];
  const float* Wk      = (const float*)d_in[20];
  const float* Wv      = (const float*)d_in[21];
  const float* Wo      = (const float*)d_in[22];
  const float* ln1_g   = (const float*)d_in[23];
  const float* ln1_b   = (const float*)d_in[24];
  const float* ln2_g   = (const float*)d_in[25];
  const float* ln2_b   = (const float*)d_in[26];
  const float* W1      = (const float*)d_in[27];
  const float* b1      = (const float*)d_in[28];
  const float* W2      = (const float*)d_in[29];
  const float* b2      = (const float*)d_in[30];
  const float* hg      = (const float*)d_in[31];
  const float* hb      = (const float*)d_in[32];
  const float* Wp1     = (const float*)d_in[33];
  const float* bp1     = (const float*)d_in[34];
  const float* Wp2     = (const float*)d_in[35];
  const float* bp2     = (const float*)d_in[36];

  // outputs: pred | pi | tau
  float* out_pred = (float*)d_out;
  float* out_pi   = out_pred + (size_t)B_ * OUT_;
  float* out_tau  = out_pi + (size_t)B_ * H_ * M_;

  // workspace layout
  char* w = (char*)d_ws;
  float*    h_raw  = (float*)(w + 0);             // B*D
  float*    h_anc  = (float*)(w + 524288);        // B*D
  float*    ss     = (float*)(w + 1048576);       // 2*D
  float*    qk     = (float*)(w + 1049088);       // D
  float*    tau_ws = (float*)(w + 1049344);       // H
  float*    ctx    = (float*)(w + 1049600);       // B*D
  float*    lg_ws  = (float*)(w + 1573888);       // B*H*K
  _Float16* hs     = (_Float16*)(w + 3671040);    // B*K*D f16

  k_anchor_gemm<<<B_ / 16, 64, 0, stream>>>(x_anc, W_anc, b_anc, h_raw);
  k_bnstats<<<1, 64, 0, stream>>>(h_raw, bn_g, bn_b, ss);
  k_bnapply<<<(B_ * D_) / 256, 256, 0, stream>>>(h_raw, ss, h_anc);
  k_prep<<<1, 64, 0, stream>>>(Wk_read, q_read, tau_raw, qk, tau_ws, out_tau);
  k_view_attn<<<B_, 256, 0, stream>>>(x_nei, ew, vmsk, W_view, b_view, Wv_read,
                                      qk, ctx, hs);
  k_gate<<<B_, 64, 0, stream>>>(g_anc, ctx, ew, vmsk, Wr1, br1, Wr2, br2,
                                out_pi, lg_ws);
  for (int l = 0; l < L_; ++l) {
    k_attn_layer<<<B_, 64, 0, stream>>>(l, hs, lg_ws, Wq, Wk, Wv, Wo,
                                        ln1_g, ln1_b, tau_ws, h_anc);
    k_ffn_layer<<<B_ / 16, 128, 0, stream>>>(l, W1, b1, W2, b2, ln2_g, ln2_b,
                                             h_anc);
  }
  k_head<<<B_, 64, 0, stream>>>(h_anc, hg, hb, Wp1, bp1, Wp2, bp2, out_pred);
}
